// FMS_61366492725323
// MI455X (gfx1250) — matrix operand order repaired and checked by reading, not compiled, not run on hardware
//
#include <hip/hip_runtime.h>

// Problem constants (from reference)
#define B_     1024
#define S_     32
#define DIM_   64
#define NREL_  237
#define CFMW_  64
#define X01LD_ 480              // [x0|x1] row stride, 474 padded to 480 (zeros)
#define XTLD_  256              // MLP input row stride, 238 padded to 256 (zeros)

typedef __attribute__((ext_vector_type(16))) _Float16 v16h;
typedef __attribute__((ext_vector_type(8)))  _Float16 v8h;
typedef __attribute__((ext_vector_type(8)))  float    v8f;

#if defined(__gfx1250__) && __has_builtin(__builtin_amdgcn_global_load_async_to_lds_b32)
#define FMS_HAS_ASYNC_LDS 1
typedef __attribute__((address_space(1))) int FMS_gint;
typedef __attribute__((address_space(3))) int FMS_lint;
#else
#define FMS_HAS_ASYNC_LDS 0
#endif

// ---------------------------------------------------------------------------
// Hop gather: M1[b*S+j, :] = (1/S) * sum_k mask * relf[r2[b,j,k], :] + relf[r1[b,j], :]
// relf (238x64, row 237 == zeros) staged in LDS via async-to-LDS;
// one wave per (b,j); per-lane neighbor index/mask broadcast via wave shuffles.
// ---------------------------------------------------------------------------
__global__ void FMS_gather_hop(const int* __restrict__ entity_pairs, int side,
                               const int* __restrict__ train_edges,
                               const int* __restrict__ entity2edges,
                               const int* __restrict__ edge2entities,
                               const int* __restrict__ edge2relation,
                               const float* __restrict__ rel_feat,
                               float* __restrict__ M1) {
  __shared__ float relf_s[(NREL_ + 1) * DIM_];
#if FMS_HAS_ASYNC_LDS
  for (int i = threadIdx.x; i < NREL_ * DIM_; i += blockDim.x) {
    __builtin_amdgcn_global_load_async_to_lds_b32(
        (FMS_gint*)(uintptr_t)(rel_feat + i),
        (FMS_lint*)(unsigned)(uintptr_t)(relf_s + i),
        0, 0);
  }
#if __has_builtin(__builtin_amdgcn_s_wait_asynccnt)
  __builtin_amdgcn_s_wait_asynccnt(0);
#else
  asm volatile("s_wait_asynccnt 0" ::: "memory");
#endif
#else
  for (int i = threadIdx.x; i < NREL_ * DIM_; i += blockDim.x) relf_s[i] = rel_feat[i];
#endif
  for (int i = threadIdx.x; i < DIM_; i += blockDim.x) relf_s[NREL_ * DIM_ + i] = 0.f;
  __syncthreads();

  int wave = blockIdx.x * (blockDim.x >> 5) + (threadIdx.x >> 5);
  int lane = threadIdx.x & 31;
  int b = wave / S_, j = wave % S_;

  int ent = entity_pairs[b * 2 + side];
  int te  = train_edges[b];
  int e0  = entity2edges[ent * S_ + j];   // wave-uniform
  int r1  = edge2relation[e0];
  int ne1 = edge2entities[e0];

  int   ek = entity2edges[ne1 * S_ + lane];   // lane == neighbor k
  float mk = (ek != te) ? 1.f : 0.f;
  int   rk = edge2relation[ek];

  float acc0 = 0.f, acc1 = 0.f;
  #pragma unroll
  for (int k = 0; k < S_; k++) {
    int   rr = __shfl(rk, k);
    float mm = __shfl(mk, k);
    acc0 += mm * relf_s[rr * DIM_ + 2 * lane];
    acc1 += mm * relf_s[rr * DIM_ + 2 * lane + 1];
  }
  const float inv = 1.f / (float)S_;
  acc0 = acc0 * inv + relf_s[r1 * DIM_ + 2 * lane];
  acc1 = acc1 * inv + relf_s[r1 * DIM_ + 2 * lane + 1];
  M1[(size_t)wave * DIM_ + 2 * lane]     = acc0;
  M1[(size_t)wave * DIM_ + 2 * lane + 1] = acc1;
}

// ---------------------------------------------------------------------------
// Second masked mean: A2[b,:] = (1/S) * sum_j mask0[b,j] * H1[b*S+j,:]
// ---------------------------------------------------------------------------
__global__ void FMS_mask_reduce(const float* __restrict__ H1,
                                const int* __restrict__ entity_pairs, int side,
                                const int* __restrict__ train_edges,
                                const int* __restrict__ entity2edges,
                                float* __restrict__ A2) {
  int b    = blockIdx.x * (blockDim.x >> 5) + (threadIdx.x >> 5);
  int lane = threadIdx.x & 31;
  int ent  = entity_pairs[b * 2 + side];
  int te   = train_edges[b];
  int e0   = entity2edges[ent * S_ + lane];   // lane == neighbor j
  float mj = (e0 != te) ? 1.f : 0.f;

  float acc0 = 0.f, acc1 = 0.f;
  #pragma unroll
  for (int j = 0; j < S_; j++) {
    float mm = __shfl(mj, j);
    const float* row = H1 + (size_t)(b * S_ + j) * DIM_;
    acc0 += mm * row[2 * lane];
    acc1 += mm * row[2 * lane + 1];
  }
  const float inv = 1.f / (float)S_;
  A2[b * DIM_ + 2 * lane]     = acc0 * inv;
  A2[b * DIM_ + 2 * lane + 1] = acc1 * inv;
}

// ---------------------------------------------------------------------------
// WMMA GEMM: C = epilogue(A(M x KPAD, f32, zero-padded cols K..KPAD-1) @
//                         W(K x N, f32) + bias)
// Block = 8 waves stacked along M over ONE 16-col N-tile (128 x 16 per block).
// Weight tile staged ONCE per block into LDS: transposed (16 x KPAD), f16,
// zero-padded for k>=K and n>=N. B fragment = two aligned 16B ds reads per
// chunk; A fragment = 4x global b128 + packed f16 converts. KPAD/ACT/MUL are
// compile-time -> constant trip count, no tails, branch-free epilogue.
// Call-site invariants: M % 128 == 0, lda == KPAD (multiple of 32).
// ACT: 0=none 1=relu 2=selu (branchless);  MUL: result *= mul (fused product)
// ---------------------------------------------------------------------------
template <int KPAD, int ACT, bool MUL>
__global__ void FMS_wmma_gemm(const float* __restrict__ A,
                              const float* __restrict__ W, int ldw,
                              const float* __restrict__ bias,
                              float* __restrict__ C, int ldc,
                              int M, int N, int K,
                              const float* __restrict__ mul, int ldmul) {
  constexpr int RSTRIDE = KPAD + 8;      // halves; multiple of 8 -> 16B rows, staggered banks
  int tilesN = (N + 15) >> 4;
  int bm = blockIdx.x / tilesN;          // block-row: 128 rows of M
  int tn = blockIdx.x % tilesN;          // N-tile
  int waveId = threadIdx.x >> 5;
  int lane   = threadIdx.x & 31;
  int tm = bm * 8 + waveId;              // this wave's M-tile (always valid)

  __shared__ __align__(16) _Float16 Wt[16 * RSTRIDE];

  // Cooperative stage: Wt[col][k] = (k<K && n<N) ? W[k][tn*16+col] : 0, as f16
  int ncol0 = tn * 16;
  for (int idx = threadIdx.x; idx < 16 * KPAD; idx += blockDim.x) {
    int col = idx & 15;
    int k   = idx >> 4;
    int nn  = ncol0 + col;
    float v = (k < K && nn < N) ? W[(size_t)k * ldw + nn] : 0.f;
    Wt[col * RSTRIDE + k] = (_Float16)v;
  }
  __syncthreads();

  int half = lane >> 4;        // 0: lanes 0-15, 1: lanes 16-31
  int l15  = lane & 15;
  int m    = tm * 16 + l15;    // A-matrix row owned by this lane (always < M)
  int n    = ncol0 + l15;      // C column owned by this lane
  int kbA  = half * 8;         // A: K halves {kbA+0..7, kbA+16..23}
  int kbB  = half * 8;         // B: K halves {kbB+0..7, kbB+16..23}
  const float* Arow = A + (size_t)m * KPAD;
  const v8h*   Wrow = (const v8h*)&Wt[l15 * RSTRIDE];   // rows 16B-aligned
  __builtin_prefetch(Arow, 0, 0);                       // global_prefetch_b8

  v8f acc = {};
  for (int kc = 0; kc < (KPAD >> 5); kc++) {
    int k0 = kc << 5;
    const float2* pa0 = (const float2*)(Arow + k0 + kbA);        // 8B-aligned
    const float2* pa1 = (const float2*)(Arow + k0 + kbA + 16);
    v16h af;
    #pragma unroll
    for (int i = 0; i < 4; i++) {
      float2 a0 = pa0[i];
      float2 a1 = pa1[i];
      af[2 * i]         = (_Float16)a0.x;
      af[2 * i + 1]     = (_Float16)a0.y;
      af[8 + 2 * i]     = (_Float16)a1.x;
      af[8 + 2 * i + 1] = (_Float16)a1.y;
    }
    // B fragment: halves K = k0+kbB .. +7 and k0+kbB+16 .. +23, two aligned 16B LDS reads
    v8h b0 = Wrow[(k0 + kbB) >> 3];
    v8h b1 = Wrow[((k0 + kbB) >> 3) + 2];
    v16h bf;
    #pragma unroll
    for (int i = 0; i < 8; i++) { bf[i] = b0[i]; bf[8 + i] = b1[i]; }

    acc = __builtin_amdgcn_wmma_f32_16x16x32_f16(false, af, false, bf,
                                                 (short)0, acc, false, false);
  }

  int ncl   = (n < N) ? n : (N - 1);
  float bv  = bias[ncl];
  int rbase = tm * 16 + half * 8;        // C: VGPR v -> row rbase+v
  if (n < N) {
    #pragma unroll
    for (int v = 0; v < 8; v++) {
      int row = rbase + v;
      float x = acc[v] + bv;
      if (ACT == 1) x = fmaxf(x, 0.f);
      if (ACT == 2) {                    // branchless selu
        const float a = 1.6732632423543772f, sc = 1.0507009873554805f;
        x = sc * fmaxf(x, 0.f) + (sc * a) * (__expf(fminf(x, 0.f)) - 1.f);
      }
      if (MUL) x *= mul[(size_t)row * ldmul + n];
      C[(size_t)row * ldc + n] = x;
    }
  }
}

// ---------------------------------------------------------------------------
// Build MLP input (and zero the pad columns of both padded A-operands):
//   XT[b, c<237]   = (1-t)*x0 + t*x1 + 0.1*eps
//   XT[b, 237]     = t
//   XT[b, 238..255]= 0
//   X01[b, 474..479] = 0
// ---------------------------------------------------------------------------
__global__ void FMS_build_xt(float* __restrict__ X01, const float* __restrict__ t,
                             const float* __restrict__ eps, float* __restrict__ XT) {
  int i = blockIdx.x * blockDim.x + threadIdx.x;
  if (i >= B_ * XTLD_) return;
  int b = i >> 8, c = i & (XTLD_ - 1);
  float tv = t[b];
  float v;
  if (c < NREL_) {
    float x0 = X01[(size_t)b * X01LD_ + c];
    float x1 = X01[(size_t)b * X01LD_ + NREL_ + c];
    v = (1.f - tv) * x0 + tv * x1 + 0.1f * eps[(size_t)b * NREL_ + c];
  } else if (c == NREL_) {
    v = tv;
  } else {
    v = 0.f;
  }
  XT[(size_t)b * XTLD_ + c] = v;
  if (c >= 238 && c < 244)                       // zero X01 cols 474..479
    X01[(size_t)b * X01LD_ + 236 + c] = 0.f;
}

// ---------------------------------------------------------------------------
extern "C" void kernel_launch(void* const* d_in, const int* in_sizes, int n_in,
                              void* d_out, int out_size, void* d_ws, size_t ws_size,
                              hipStream_t stream) {
  const int*   ep    = (const int*)d_in[0];
  const int*   te    = (const int*)d_in[1];
  // d_in[2] = labels: dead in the reference dataflow (only feeds the unused ev0' branch)
  const int*   e2e   = (const int*)d_in[3];
  const int*   eent  = (const int*)d_in[4];
  const int*   erel  = (const int*)d_in[5];
  const float* t     = (const float*)d_in[6];
  const float* eps   = (const float*)d_in[7];
  const float* relf  = (const float*)d_in[8];
  const float* Wagg0 = (const float*)d_in[9];
  const float* bagg0 = (const float*)d_in[10];
  const float* Wagg1 = (const float*)d_in[11];
  const float* bagg1 = (const float*)d_in[12];
  const float* Wout  = (const float*)d_in[13];
  const float* bout  = (const float*)d_in[14];
  const float* mW1 = (const float*)d_in[15]; const float* mb1 = (const float*)d_in[16];
  const float* mW2 = (const float*)d_in[17]; const float* mb2 = (const float*)d_in[18];
  const float* mW3 = (const float*)d_in[19]; const float* mb3 = (const float*)d_in[20];
  const float* mW4 = (const float*)d_in[21]; const float* mb4 = (const float*)d_in[22];

  // Workspace layout (floats)
  float* ws  = (float*)d_ws;
  float* M1  = ws;                                  // 32768*64
  float* H1  = M1  + (size_t)B_ * S_ * DIM_;        // 32768*64
  float* A2  = H1  + (size_t)B_ * S_ * DIM_;        // 1024*64
  float* X01 = A2  + (size_t)B_ * DIM_;             // 1024*480 ([x0|x1], padded)
  float* XT  = X01 + (size_t)B_ * X01LD_;           // 1024*256 (padded)
  float* Hm1 = XT  + (size_t)B_ * XTLD_;            // 1024*64
  float* Hm2 = Hm1 + (size_t)B_ * CFMW_;            // 1024*64
  float* Hm3 = Hm2 + (size_t)B_ * CFMW_;            // 1024*64
  float* OUTb = Hm3 + (size_t)B_ * CFMW_;           // 1024*237
  size_t need = ((size_t)(OUTb - ws) + (size_t)B_ * NREL_) * sizeof(float);
  if (ws_size < need) return;

  auto gblocks = [](int M, int N) {
    return (M / 128) * ((N + 15) / 16);             // 8 waves/block, stacked along M
  };

  for (int s = 0; s < 2; s++) {
    FMS_gather_hop<<<(B_ * S_) / 8, 256, 0, stream>>>(ep, s, te, e2e, eent, erel, relf, M1);
    // H1 = relu(M1 @ W_agg0 + b_agg0)         (32768 x 64 x 64)
    FMS_wmma_gemm<64, 1, false><<<gblocks(B_ * S_, DIM_), 256, 0, stream>>>(
        M1, Wagg0, DIM_, bagg0, H1, DIM_, B_ * S_, DIM_, DIM_, nullptr, 0);
    FMS_mask_reduce<<<B_ / 8, 256, 0, stream>>>(H1, ep, s, te, e2e, A2);
    // x_s = A2 @ W_agg1 + b_agg1  -> columns [s*237, s*237+237) of X01
    FMS_wmma_gemm<64, 0, false><<<gblocks(B_, NREL_), 256, 0, stream>>>(
        A2, Wagg1, NREL_, bagg1, X01 + s * NREL_, X01LD_,
        B_, NREL_, DIM_, nullptr, 0);
  }

  FMS_build_xt<<<(B_ * XTLD_) / 256, 256, 0, stream>>>(X01, t, eps, XT);

  // output = [x0|x1] @ W_out + b_out          (1024 x 237 x 474, KPAD 480)
  FMS_wmma_gemm<X01LD_, 0, false><<<gblocks(B_, NREL_), 256, 0, stream>>>(
      X01, Wout, NREL_, bout, OUTb, NREL_, B_, NREL_, 2 * NREL_, nullptr, 0);

  // SELU MLP chain
  FMS_wmma_gemm<XTLD_, 2, false><<<gblocks(B_, CFMW_), 256, 0, stream>>>(
      XT, mW1, CFMW_, mb1, Hm1, CFMW_, B_, CFMW_, NREL_ + 1, nullptr, 0);
  FMS_wmma_gemm<64, 2, false><<<gblocks(B_, CFMW_), 256, 0, stream>>>(
      Hm1, mW2, CFMW_, mb2, Hm2, CFMW_, B_, CFMW_, CFMW_, nullptr, 0);
  FMS_wmma_gemm<64, 2, false><<<gblocks(B_, CFMW_), 256, 0, stream>>>(
      Hm2, mW3, CFMW_, mb3, Hm3, CFMW_, B_, CFMW_, CFMW_, nullptr, 0);

  // scores = (Hm3 @ mW4 + mb4) * output  -- fused multiply epilogue, writes d_out
  FMS_wmma_gemm<64, 0, true><<<gblocks(B_, NREL_), 256, 0, stream>>>(
      Hm3, mW4, NREL_, mb4, (float*)d_out, NREL_, B_, NREL_, CFMW_, OUTb, NREL_);
}
